// PredictedLDDTHead_18176301597146
// MI455X (gfx1250) — compile-verified, matrix operand order checked by reading, not run on hardware
//
#include <hip/hip_runtime.h>

typedef __attribute__((ext_vector_type(16))) __bf16       v16bf;
typedef __attribute__((ext_vector_type(8)))  float        v8f;
typedef __attribute__((ext_vector_type(8)))  unsigned int v8u;

#define N_RES   262144
#define C_S     384
#define C_H     128
#define N_BINS  50
#define NB_PAD  64

#define KT0 12
#define NT0 8
#define KT1 4
#define NT1 8
#define KT2 4
#define NT2 4

#define WAVES   8
#define ROWS_PER_WAVE 16
#define ROWS_PER_BLOCK (WAVES * ROWS_PER_WAVE)   // 128

// ---- prepped weight image layout in d_ws (bytes) ----
#define PW0_OFF   0        // 12*8 frags * 1024B        = 98304
#define PW1_OFF   98304    // 4*8 frags                 = 32768
#define PW2_OFF   131072   // 4*4 frags                 = 16384
#define PPAR_OFF  147456   // floats: lnS[384] lnB[384] b0[128] b1[128] b2pad[64]
#define PREP_BYTES 151808  // = 37*4096 + 256
#define FULL_SWEEPS 37

#define SACT_OFF  151808               // per-wave A scratch: 8 * 4KB
#define LDS_BYTES (SACT_OFF + WAVES * KT1 * 512 * 2)   // 184576 B

// ---------- bf16 helpers ----------
static __device__ __forceinline__ unsigned int f2bf_rne(float f) {   // exact RNE (prep only)
    unsigned int u = __builtin_bit_cast(unsigned int, f);
    return (u + 0x7fffu + ((u >> 16) & 1u)) >> 16;
}
// fast path: round-to-nearest (+0x8000) then byte-perm packs a pair in 3 VALU
static __device__ __forceinline__ unsigned int pack_bf2_rn(float lo, float hi) {
    unsigned int a = __builtin_bit_cast(unsigned int, lo) + 0x8000u;
    unsigned int b = __builtin_bit_cast(unsigned int, hi) + 0x8000u;
    return __builtin_amdgcn_perm(b, a, 0x07060302);   // [hi16(b) : hi16(a)]
}
static __device__ __forceinline__ unsigned short f2bf_rn(float f) {
    return (unsigned short)((__builtin_bit_cast(unsigned int, f) + 0x8000u) >> 16);
}
static __device__ __forceinline__ v16bf as_bf16x16(v8u u) {
    return __builtin_bit_cast(v16bf, u);
}

// fragment swizzle: element (k'=k%32, n'=n%16) of a 16x32 bf16 WMMA operand
static __device__ __forceinline__ int frag_lane(int kp, int np) {
    return np + (((kp >> 3) & 1) << 4);
}
static __device__ __forceinline__ int frag_idx(int kp) {
    return (kp & 7) + (((kp >> 4) & 1) << 3);
}

// wave-local LDS handoff: compiler barrier + wait for own DS ops (DS is in-order per wave)
static __device__ __forceinline__ void wave_lds_fence() {
    asm volatile("s_wait_dscnt 0x0" ::: "memory");
}

// ================= prep kernel: swizzle weights -> fragment image in d_ws ===============
__global__ void prep_weights(const float* __restrict__ w0, const float* __restrict__ w1,
                             const float* __restrict__ w2, const float* __restrict__ b0,
                             const float* __restrict__ b1, const float* __restrict__ b2,
                             const float* __restrict__ lnS, const float* __restrict__ lnB,
                             unsigned char* __restrict__ ws)
{
    const int tid0   = blockIdx.x * blockDim.x + threadIdx.x;
    const int stride = gridDim.x * blockDim.x;
    unsigned short* W0 = (unsigned short*)(ws + PW0_OFF);
    unsigned short* W1 = (unsigned short*)(ws + PW1_OFF);
    unsigned short* W2 = (unsigned short*)(ws + PW2_OFF);
    float*          P  = (float*)(ws + PPAR_OFF);

    for (int i = tid0; i < C_S * C_H; i += stride) {
        int k = i >> 7, n = i & 127;
        int kt = k >> 5, nt = n >> 4, kp = k & 31, np = n & 15;
        W0[(kt * NT0 + nt) * 512 + frag_lane(kp, np) * 16 + frag_idx(kp)] =
            (unsigned short)f2bf_rne(w0[i]);
    }
    for (int i = tid0; i < C_H * C_H; i += stride) {
        int k = i >> 7, n = i & 127;
        int kt = k >> 5, nt = n >> 4, kp = k & 31, np = n & 15;
        W1[(kt * NT1 + nt) * 512 + frag_lane(kp, np) * 16 + frag_idx(kp)] =
            (unsigned short)f2bf_rne(w1[i]);
    }
    for (int i = tid0; i < C_H * NB_PAD; i += stride) {
        int k = i >> 6, n = i & 63;
        float v = (n < N_BINS) ? w2[k * N_BINS + n] : 0.0f;
        int kt = k >> 5, nt = n >> 4, kp = k & 31, np = n & 15;
        W2[(kt * NT2 + nt) * 512 + frag_lane(kp, np) * 16 + frag_idx(kp)] =
            (unsigned short)f2bf_rne(v);
    }
    for (int i = tid0; i < C_S; i += stride) { P[i] = lnS[i]; P[384 + i] = lnB[i]; }
    for (int i = tid0; i < C_H; i += stride) { P[768 + i] = b0[i]; P[896 + i] = b1[i]; }
    for (int i = tid0; i < NB_PAD; i += stride) P[1024 + i] = (i < N_BINS) ? b2[i] : 0.0f;
}

// ================= main fused kernel =================
__global__ __launch_bounds__(256, 1)
void lddt_head_fused(const float* __restrict__ x,
                     const unsigned char* __restrict__ prep,
                     float* __restrict__ out)
{
    __shared__ __align__(32) unsigned char sMem[LDS_BYTES];
    unsigned short* sW0  = (unsigned short*)(sMem + PW0_OFF);
    unsigned short* sW1  = (unsigned short*)(sMem + PW1_OFF);
    unsigned short* sW2  = (unsigned short*)(sMem + PW2_OFF);
    const float*    sLnS = (const float*)(sMem + PPAR_OFF);
    const float*    sLnB = sLnS + 384;
    const float*    sB0  = sLnS + 768;
    const float*    sB1  = sLnS + 896;
    const float*    sB2  = sLnS + 1024;

    const int tid  = threadIdx.x;
    const int wave = tid >> 5;
    const int lane = tid & 31;
    const int mrow = lane & 15;
    const int half = lane >> 4;

    // -------- kick off async LDS fill of the pre-swizzled weight image --------
    {
        const unsigned ldsbase = (unsigned)(uintptr_t)&sMem[0];
        const unsigned long long gbase = (unsigned long long)(uintptr_t)prep;
        for (int s2 = 0; s2 < FULL_SWEEPS; ++s2) {
            unsigned off = (unsigned)(s2 * 4096) + (unsigned)tid * 16u;
            asm volatile("global_load_async_to_lds_b128 %0, %1, %2"
                         :: "v"(ldsbase + off), "v"(off), "s"(gbase) : "memory");
        }
        unsigned off = (unsigned)(FULL_SWEEPS * 4096) + (unsigned)tid * 16u;
        if (off < PREP_BYTES) {
            asm volatile("global_load_async_to_lds_b128 %0, %1, %2"
                         :: "v"(ldsbase + off), "v"(off), "s"(gbase) : "memory");
        }
    }

    // -------- pass 1: LayerNorm statistics (overlaps the async weight copy) --------
    const long row0 = ((long)blockIdx.x * WAVES + wave) * ROWS_PER_WAVE;
    const float* xrow = x + (row0 + mrow) * C_S;

    float sum = 0.0f, ssq = 0.0f;
    #pragma unroll
    for (int kt = 0; kt < KT0; ++kt) {
        const float4* p1 = (const float4*)(xrow + kt * 32 + half * 8);
        const float4* p2 = (const float4*)(xrow + kt * 32 + 16 + half * 8);
        float4 f0 = p1[0], f1 = p1[1], f2 = p2[0], f3 = p2[1];
        sum += f0.x + f0.y + f0.z + f0.w + f1.x + f1.y + f1.z + f1.w
             + f2.x + f2.y + f2.z + f2.w + f3.x + f3.y + f3.z + f3.w;
        ssq += f0.x*f0.x + f0.y*f0.y + f0.z*f0.z + f0.w*f0.w
             + f1.x*f1.x + f1.y*f1.y + f1.z*f1.z + f1.w*f1.w
             + f2.x*f2.x + f2.y*f2.y + f2.z*f2.z + f2.w*f2.w
             + f3.x*f3.x + f3.y*f3.y + f3.z*f3.z + f3.w*f3.w;
    }
    sum += __shfl_xor(sum, 16, 32);
    ssq += __shfl_xor(ssq, 16, 32);
    const float mu  = sum * (1.0f / (float)C_S);
    const float var = ssq * (1.0f / (float)C_S) - mu * mu;
    const float rs  = rsqrtf(var + 1e-5f);

    // weights + LN params must be resident before pass 2
    asm volatile("s_wait_asynccnt 0x0" ::: "memory");
    __syncthreads();

    // -------- pass 2: reload x (L2-hot), normalize fp32, pack A0 fragments --------
    v8u a0[KT0];
    #pragma unroll
    for (int kt = 0; kt < KT0; ++kt) {
        const int c1 = kt * 32 + half * 8;
        const int c2 = c1 + 16;
        const float4* p1 = (const float4*)(xrow + c1);
        const float4* p2 = (const float4*)(xrow + c2);
        float4 f0 = p1[0], f1 = p1[1], f2 = p2[0], f3 = p2[1];
        float4 g0 = *(const float4*)&sLnS[c1], g1 = *(const float4*)&sLnS[c1 + 4];
        float4 g2 = *(const float4*)&sLnS[c2], g3 = *(const float4*)&sLnS[c2 + 4];
        float4 h0 = *(const float4*)&sLnB[c1], h1 = *(const float4*)&sLnB[c1 + 4];
        float4 h2 = *(const float4*)&sLnB[c2], h3 = *(const float4*)&sLnB[c2 + 4];
        a0[kt][0] = pack_bf2_rn((f0.x - mu) * (rs * g0.x) + h0.x,
                                (f0.y - mu) * (rs * g0.y) + h0.y);
        a0[kt][1] = pack_bf2_rn((f0.z - mu) * (rs * g0.z) + h0.z,
                                (f0.w - mu) * (rs * g0.w) + h0.w);
        a0[kt][2] = pack_bf2_rn((f1.x - mu) * (rs * g1.x) + h1.x,
                                (f1.y - mu) * (rs * g1.y) + h1.y);
        a0[kt][3] = pack_bf2_rn((f1.z - mu) * (rs * g1.z) + h1.z,
                                (f1.w - mu) * (rs * g1.w) + h1.w);
        a0[kt][4] = pack_bf2_rn((f2.x - mu) * (rs * g2.x) + h2.x,
                                (f2.y - mu) * (rs * g2.y) + h2.y);
        a0[kt][5] = pack_bf2_rn((f2.z - mu) * (rs * g2.z) + h2.z,
                                (f2.w - mu) * (rs * g2.w) + h2.w);
        a0[kt][6] = pack_bf2_rn((f3.x - mu) * (rs * g3.x) + h3.x,
                                (f3.y - mu) * (rs * g3.y) + h3.y);
        a0[kt][7] = pack_bf2_rn((f3.z - mu) * (rs * g3.z) + h3.z,
                                (f3.w - mu) * (rs * g3.w) + h3.w);
    }

    unsigned short* act = (unsigned short*)(sMem + SACT_OFF) + wave * (KT1 * 512);

    // -------- GEMM0: [16x384]x[384x128] + b0, ReLU -> A1 fragments (per-wave LDS) ------
    #pragma unroll
    for (int nt = 0; nt < NT0; ++nt) {
        const float bias = sB0[nt * 16 + mrow];
        v8f acc;
        #pragma unroll
        for (int r = 0; r < 8; ++r) acc[r] = bias;
        #pragma unroll
        for (int kt = 0; kt < KT0; ++kt) {
            v16bf a = as_bf16x16(a0[kt]);
            v16bf b = as_bf16x16(*(const v8u*)&sW0[(kt * NT0 + nt) * 512 + lane * 16]);
            acc = __builtin_amdgcn_wmma_f32_16x16x32_bf16(false, a, false, b,
                                                          (short)0, acc, false, false);
        }
        #pragma unroll
        for (int r = 0; r < 8; ++r) {
            float v = acc[r]; v = v > 0.0f ? v : 0.0f;
            int m = r + half * 8;
            int c = nt * 16 + mrow;
            int kt1 = c >> 5, cp = c & 31;
            act[kt1 * 512 + frag_lane(cp, m) * 16 + frag_idx(cp)] = f2bf_rn(v);
        }
    }
    wave_lds_fence();

    // -------- GEMM1: [16x128]x[128x128] + b1, ReLU -> A2 fragments --------
    v8u a1[KT1];
    #pragma unroll
    for (int kt = 0; kt < KT1; ++kt)
        a1[kt] = *(const v8u*)&act[kt * 512 + lane * 16];
    wave_lds_fence();   // reads done before scratch is overwritten

    #pragma unroll
    for (int nt = 0; nt < NT1; ++nt) {
        const float bias = sB1[nt * 16 + mrow];
        v8f acc;
        #pragma unroll
        for (int r = 0; r < 8; ++r) acc[r] = bias;
        #pragma unroll
        for (int kt = 0; kt < KT1; ++kt) {
            v16bf a = as_bf16x16(a1[kt]);
            v16bf b = as_bf16x16(*(const v8u*)&sW1[(kt * NT1 + nt) * 512 + lane * 16]);
            acc = __builtin_amdgcn_wmma_f32_16x16x32_bf16(false, a, false, b,
                                                          (short)0, acc, false, false);
        }
        #pragma unroll
        for (int r = 0; r < 8; ++r) {
            float v = acc[r]; v = v > 0.0f ? v : 0.0f;
            int m = r + half * 8;
            int c = nt * 16 + mrow;
            int kt1 = c >> 5, cp = c & 31;
            act[kt1 * 512 + frag_lane(cp, m) * 16 + frag_idx(cp)] = f2bf_rn(v);
        }
    }
    wave_lds_fence();

    // -------- GEMM2: [16x128]x[128x64(pad)] + b2 -> masked fp32 stores --------
    v8u a2[KT2];
    #pragma unroll
    for (int kt = 0; kt < KT2; ++kt)
        a2[kt] = *(const v8u*)&act[kt * 512 + lane * 16];

    #pragma unroll
    for (int nt = 0; nt < NT2; ++nt) {
        const float bias = sB2[nt * 16 + mrow];
        v8f acc;
        #pragma unroll
        for (int r = 0; r < 8; ++r) acc[r] = bias;
        #pragma unroll
        for (int kt = 0; kt < KT2; ++kt) {
            v16bf a = as_bf16x16(a2[kt]);
            v16bf b = as_bf16x16(*(const v8u*)&sW2[(kt * NT2 + nt) * 512 + lane * 16]);
            acc = __builtin_amdgcn_wmma_f32_16x16x32_bf16(false, a, false, b,
                                                          (short)0, acc, false, false);
        }
        const int n = nt * 16 + mrow;
        if (n < N_BINS) {
            #pragma unroll
            for (int r = 0; r < 8; ++r) {
                long orow = row0 + r + half * 8;
                out[orow * N_BINS + n] = acc[r];
            }
        }
    }
}

extern "C" void kernel_launch(void* const* d_in, const int* in_sizes, int n_in,
                              void* d_out, int out_size, void* d_ws, size_t ws_size,
                              hipStream_t stream) {
    (void)in_sizes; (void)n_in; (void)out_size; (void)ws_size;
    const float* x   = (const float*)d_in[0];
    const float* lns = (const float*)d_in[1];
    const float* lnb = (const float*)d_in[2];
    const float* w0  = (const float*)d_in[3];
    const float* b0  = (const float*)d_in[4];
    const float* w1  = (const float*)d_in[5];
    const float* b1  = (const float*)d_in[6];
    const float* w2  = (const float*)d_in[7];
    const float* b2  = (const float*)d_in[8];
    float* out = (float*)d_out;

    hipLaunchKernelGGL(prep_weights, dim3(64), dim3(256), 0, stream,
                       w0, w1, w2, b0, b1, b2, lns, lnb, (unsigned char*)d_ws);
    hipLaunchKernelGGL(lddt_head_fused, dim3(N_RES / ROWS_PER_BLOCK), dim3(256), 0, stream,
                       x, (const unsigned char*)d_ws, out);
}
